// GCN_86260123174489
// MI455X (gfx1250) — matrix operand order repaired and checked by reading, not compiled, not run on hardware
//
#include <hip/hip_runtime.h>
#include <hip/hip_bf16.h>

typedef __attribute__((ext_vector_type(16))) __bf16 v16bf;
typedef __attribute__((ext_vector_type(8)))  float  v8f;

#define WAVES_PER_BLOCK 8

// ---------------------------------------------------------------------------
// Degree / normalization
// ---------------------------------------------------------------------------
__global__ void deg_init(float* __restrict__ deg, int n) {
    int i = blockIdx.x * blockDim.x + threadIdx.x;
    if (i < n) deg[i] = 1.0f;                      // self-loop
}

__global__ void deg_accum(const long long* __restrict__ ei, float* __restrict__ deg, int E) {
    int e = blockIdx.x * blockDim.x + threadIdx.x;
    if (e < E) atomicAdd(&deg[(int)ei[E + e]], 1.0f);
}

__global__ void deg_rsqrt(float* __restrict__ deg, int n) {
    int i = blockIdx.x * blockDim.x + threadIdx.x;
    if (i < n) deg[i] = rsqrtf(deg[i]);
}

// ---------------------------------------------------------------------------
// fp32 -> bf16 conversion / weight transpose
// ---------------------------------------------------------------------------
__global__ void f32_to_bf16(const float* __restrict__ src, __hip_bfloat16* __restrict__ dst,
                            long long n) {
    long long i = (long long)blockIdx.x * blockDim.x + threadIdx.x;
    if (i < n) dst[i] = __float2bfloat16(src[i]);
}

// W: K x Ncols row-major fp32  ->  Wt: Ncols x K row-major bf16
__global__ void conv_transpose_bf16(const float* __restrict__ W, __hip_bfloat16* __restrict__ Wt,
                                    int K, int Ncols) {
    int i = blockIdx.x * blockDim.x + threadIdx.x;
    if (i >= K * Ncols) return;
    int k = i / Ncols;
    int n = i - k * Ncols;
    Wt[(size_t)n * K + k] = __float2bfloat16(W[i]);
}

// ---------------------------------------------------------------------------
// WMMA GEMM:  C[M x Ncols] (f32) = A[M x K] (bf16, row-major) @ Bt^T
//             Bt is Ncols x K row-major bf16 (i.e. B transposed)
// One wave -> 16x64 output tile (4 wmma accumulators). M%16==0, Ncols%64==0.
// ---------------------------------------------------------------------------
__global__ __launch_bounds__(256) void gcn_gemm_bf16_wmma(
    const __hip_bfloat16* __restrict__ A,
    const __hip_bfloat16* __restrict__ Bt,
    float* __restrict__ C,
    int M, int K, int Ncols)
{
    const int lane = threadIdx.x & 31;
    const int wave = threadIdx.y;
    const int rowTiles  = M >> 4;
    const int colGroups = Ncols >> 6;
    const int tile = blockIdx.x * WAVES_PER_BLOCK + wave;
    if (tile >= rowTiles * colGroups) return;
    const int rowTile  = tile / colGroups;
    const int colGroup = tile - rowTile * colGroups;

    const int n    = lane & 15;   // A row within tile / B column within tile
    const int half = lane >> 4;

    const __hip_bfloat16* arow = A + (size_t)(rowTile * 16 + n) * K;
    const int col0 = colGroup * 64;
    const __hip_bfloat16* b0r = Bt + (size_t)(col0 +  0 + n) * K;
    const __hip_bfloat16* b1r = Bt + (size_t)(col0 + 16 + n) * K;
    const __hip_bfloat16* b2r = Bt + (size_t)(col0 + 32 + n) * K;
    const __hip_bfloat16* b3r = Bt + (size_t)(col0 + 48 + n) * K;

    const int aoff  = half * 16;   // A: K offset of this lane's 16 contiguous elems
    const int kboff = half * 16;   // B: K offset of this lane's 16 contiguous elems

    v8f acc0 = {}, acc1 = {}, acc2 = {}, acc3 = {};

    for (int k0 = 0; k0 < K; k0 += 32) {
        // A fragment: 16 contiguous bf16 (K=[aoff,aoff+16))
        union { uint4 u[2]; v16bf v; } af;
        const uint4* ap = (const uint4*)(arow + k0 + aoff);
        af.u[0] = ap[0];
        af.u[1] = ap[1];                      // +8 bf16 elements = +1 uint4

        // B fragments: 16 contiguous bf16 per lane from transposed weights
        union { uint4 u[2]; v16bf v; } bf0, bf1, bf2, bf3;
        const uint4* p0 = (const uint4*)(b0r + k0 + kboff);
        const uint4* p1 = (const uint4*)(b1r + k0 + kboff);
        const uint4* p2 = (const uint4*)(b2r + k0 + kboff);
        const uint4* p3 = (const uint4*)(b3r + k0 + kboff);
        bf0.u[0] = p0[0]; bf0.u[1] = p0[1];
        bf1.u[0] = p1[0]; bf1.u[1] = p1[1];
        bf2.u[0] = p2[0]; bf2.u[1] = p2[1];
        bf3.u[0] = p3[0]; bf3.u[1] = p3[1];

        acc0 = __builtin_amdgcn_wmma_f32_16x16x32_bf16(false, af.v, false, bf0.v,
                                                       (short)0, acc0, false, false);
        acc1 = __builtin_amdgcn_wmma_f32_16x16x32_bf16(false, af.v, false, bf1.v,
                                                       (short)0, acc1, false, false);
        acc2 = __builtin_amdgcn_wmma_f32_16x16x32_bf16(false, af.v, false, bf2.v,
                                                       (short)0, acc2, false, false);
        acc3 = __builtin_amdgcn_wmma_f32_16x16x32_bf16(false, af.v, false, bf3.v,
                                                       (short)0, acc3, false, false);
    }

    // C/D layout: VGPR v -> row (v + 8*half), col n
    float* cbase = C + (size_t)(rowTile * 16 + half * 8) * Ncols + col0 + n;
    #pragma unroll
    for (int v = 0; v < 8; ++v) {
        size_t r = (size_t)v * Ncols;
        cbase[r +  0] = acc0[v];
        cbase[r + 16] = acc1[v];
        cbase[r + 32] = acc2[v];
        cbase[r + 48] = acc3[v];
    }
}

// ---------------------------------------------------------------------------
// Aggregation: Agg[dst] += dinv[src]*dinv[dst] * G[src]
// Edges e in [0,E) from edge_index; e in [E, E+Nn) are self-loops.
// One thread = one (edge, float4 chunk).  chunkShift = log2(Hc/4).
// ---------------------------------------------------------------------------
__global__ void gcn_aggregate(const long long* __restrict__ ei,
                              const float* __restrict__ dinv,
                              const float* __restrict__ G,
                              float* __restrict__ Agg,
                              int E, int Nn, int Hc, int chunkShift)
{
    long long idx = (long long)blockIdx.x * blockDim.x + threadIdx.x;
    long long total = (long long)(E + Nn) << chunkShift;
    if (idx >= total) return;
    int e = (int)(idx >> chunkShift);
    int c = ((int)idx & ((1 << chunkShift) - 1)) << 2;
    int s, d;
    if (e < E) { s = (int)ei[e]; d = (int)ei[E + e]; }
    else       { s = e - E; d = s; }
    float nrm = dinv[s] * dinv[d];
    const float4 g = *(const float4*)(G + (size_t)s * Hc + c);
    float* o = Agg + (size_t)d * Hc + c;
    atomicAdd(o + 0, g.x * nrm);
    atomicAdd(o + 1, g.y * nrm);
    atomicAdd(o + 2, g.z * nrm);
    atomicAdd(o + 3, g.w * nrm);
}

// ---------------------------------------------------------------------------
// Epilogues
// ---------------------------------------------------------------------------
__global__ void zero_f32(float* __restrict__ p, int n) {
    int i = blockIdx.x * blockDim.x + threadIdx.x;
    if (i < n) p[i] = 0.0f;
}

// acc = elu(acc + bias [+ res]); also emit bf16 copy for the next GEMM
__global__ void epilogue_elu(float* __restrict__ acc, const float* __restrict__ bias,
                             const float* __restrict__ res, __hip_bfloat16* __restrict__ obf,
                             int total, int mask)
{
    int i = blockIdx.x * blockDim.x + threadIdx.x;
    if (i >= total) return;
    float v = acc[i] + bias[i & mask];
    if (res) v += res[i];
    v = (v > 0.0f) ? v : (expf(v) - 1.0f);
    acc[i] = v;
    obf[i] = __float2bfloat16(v);
}

__global__ void bias_add(float* __restrict__ out, const float* __restrict__ bias,
                         int total, int mask)
{
    int i = blockIdx.x * blockDim.x + threadIdx.x;
    if (i < total) out[i] += bias[i & mask];
}

// ---------------------------------------------------------------------------
// Launch
// ---------------------------------------------------------------------------
extern "C" void kernel_launch(void* const* d_in, const int* in_sizes, int n_in,
                              void* d_out, int out_size, void* d_ws, size_t ws_size,
                              hipStream_t stream)
{
    const float*     x  = (const float*)d_in[0];
    const long long* ei = (const long long*)d_in[1];
    const float*     W1 = (const float*)d_in[2];
    const float*     b1 = (const float*)d_in[3];
    const float*     W2 = (const float*)d_in[4];
    const float*     b2 = (const float*)d_in[5];
    const float*     W3 = (const float*)d_in[6];
    const float*     b3 = (const float*)d_in[7];
    float* out = (float*)d_out;

    const int F = 512, H = 256, Cc = 64;
    const int Nn = in_sizes[0] / F;      // 50000
    const int E  = in_sizes[1] / 2;      // 800000
    (void)n_in; (void)out_size; (void)ws_size;

    char* ws = (char*)d_ws;
    size_t off = 0;
    auto alloc = [&](size_t bytes) {
        size_t o = off; off += (bytes + 255) & ~(size_t)255; return o;
    };
    float*          dinv = (float*)         (ws + alloc((size_t)Nn * 4));
    __hip_bfloat16* Wt1  = (__hip_bfloat16*)(ws + alloc((size_t)H  * F * 2));
    __hip_bfloat16* Wt2  = (__hip_bfloat16*)(ws + alloc((size_t)H  * H * 2));
    __hip_bfloat16* Wt3  = (__hip_bfloat16*)(ws + alloc((size_t)Cc * H * 2));
    __hip_bfloat16* Xbf  = (__hip_bfloat16*)(ws + alloc((size_t)Nn * F * 2));
    float*          G    = (float*)         (ws + alloc((size_t)Nn * H * 4));
    float*          Agg1 = (float*)         (ws + alloc((size_t)Nn * H * 4));
    float*          Agg2 = (float*)         (ws + alloc((size_t)Nn * H * 4));
    // X region recycled: x1bf in first half, x2bf in second half
    __hip_bfloat16* x1bf = Xbf;
    __hip_bfloat16* x2bf = Xbf + (size_t)Nn * H;

    // --- degree & dinv ---
    deg_init <<<(Nn + 255) / 256, 256, 0, stream>>>(dinv, Nn);
    deg_accum<<<(E  + 255) / 256, 256, 0, stream>>>(ei, dinv, E);
    deg_rsqrt<<<(Nn + 255) / 256, 256, 0, stream>>>(dinv, Nn);

    // --- conversions ---
    long long nx = (long long)Nn * F;
    f32_to_bf16<<<(int)((nx + 255) / 256), 256, 0, stream>>>(x, Xbf, nx);
    conv_transpose_bf16<<<(F * H  + 255) / 256, 256, 0, stream>>>(W1, Wt1, F, H);
    conv_transpose_bf16<<<(H * H  + 255) / 256, 256, 0, stream>>>(W2, Wt2, H, H);
    conv_transpose_bf16<<<(H * Cc + 255) / 256, 256, 0, stream>>>(W3, Wt3, H, Cc);

    dim3 gthr(32, WAVES_PER_BLOCK);
    auto gemm_blocks = [&](int M, int Ncols) {
        int t = (M >> 4) * (Ncols >> 6);
        return (t + WAVES_PER_BLOCK - 1) / WAVES_PER_BLOCK;
    };

    // ---------------- Layer 1 ----------------
    gcn_gemm_bf16_wmma<<<gemm_blocks(Nn, H), gthr, 0, stream>>>(Xbf, Wt1, G, Nn, F, H);
    zero_f32<<<(Nn * H + 255) / 256, 256, 0, stream>>>(Agg1, Nn * H);
    {
        long long tot = (long long)(E + Nn) << 6;   // H/4 = 64 chunks
        gcn_aggregate<<<(int)((tot + 255) / 256), 256, 0, stream>>>(ei, dinv, G, Agg1,
                                                                    E, Nn, H, 6);
    }
    epilogue_elu<<<(Nn * H + 255) / 256, 256, 0, stream>>>(Agg1, b1, nullptr, x1bf,
                                                           Nn * H, H - 1);

    // ---------------- Layer 2 ----------------
    gcn_gemm_bf16_wmma<<<gemm_blocks(Nn, H), gthr, 0, stream>>>(x1bf, Wt2, G, Nn, H, H);
    zero_f32<<<(Nn * H + 255) / 256, 256, 0, stream>>>(Agg2, Nn * H);
    {
        long long tot = (long long)(E + Nn) << 6;
        gcn_aggregate<<<(int)((tot + 255) / 256), 256, 0, stream>>>(ei, dinv, G, Agg2,
                                                                    E, Nn, H, 6);
    }
    epilogue_elu<<<(Nn * H + 255) / 256, 256, 0, stream>>>(Agg2, b2, Agg1 /*residual x1*/,
                                                           x2bf, Nn * H, H - 1);

    // ---------------- Layer 3 ----------------
    gcn_gemm_bf16_wmma<<<gemm_blocks(Nn, Cc), gthr, 0, stream>>>(x2bf, Wt3, G, Nn, H, Cc);
    zero_f32<<<(Nn * Cc + 255) / 256, 256, 0, stream>>>(out, Nn * Cc);
    {
        long long tot = (long long)(E + Nn) << 4;   // C/4 = 16 chunks
        gcn_aggregate<<<(int)((tot + 255) / 256), 256, 0, stream>>>(ei, dinv, G, out,
                                                                    E, Nn, Cc, 4);
    }
    bias_add<<<(Nn * Cc + 255) / 256, 256, 0, stream>>>(out, b3, Nn * Cc, Cc - 1);
}
